// ClusteringLayer_34050500723295
// MI455X (gfx1250) — matrix operand order repaired and checked by reading, not compiled, not run on hardware
//
#include <hip/hip_runtime.h>

// ---------------- CDNA5 types ----------------
typedef __attribute__((ext_vector_type(16))) __bf16 v16bf;
typedef __attribute__((ext_vector_type(8)))  float  v8f;

// ---------------- problem constants ----------------
constexpr int ND = 512;    // feature dim D
constexpr int NK = 1024;   // clusters K
constexpr int BM = 64;     // rows per workgroup
constexpr int THREADS = 512;            // 16 waves
constexpr int DSTEPS = ND / 32;         // 16 WMMA K-steps
constexpr int MT = 4;                   // 4 row tiles (all) per wave
constexpr int KT = 4;                   // 4 col tiles (64 cols) per wave

union FragAB {
  v16bf    v;
  uint4    q[2];
  unsigned u[8];
};

// fp32 -> bf16 pair, round-half-up + single v_perm_b32 pack (3 VALU total)
__device__ __forceinline__ unsigned pk_bf16(float a, float b) {
  unsigned ua = __float_as_uint(a) + 0x8000u;
  unsigned ub = __float_as_uint(b) + 0x8000u;
  return __builtin_amdgcn_perm(ub, ua, 0x07060302u);
}

// ---------------------------------------------------------------------------
// Kernel 1: clusters fp32 -> bf16 into workspace, plus ||c||^2 per cluster.
// 2 MB read / 1 MB write: negligible next to the main kernel.
// ---------------------------------------------------------------------------
__global__ __launch_bounds__(256)
void prep_clusters(const float* __restrict__ clusters,
                   unsigned short* __restrict__ cbf,
                   float* __restrict__ csqg) {
  __shared__ float partial[256];
  const int row = blockIdx.x;     // one cluster row per block
  const int t   = threadIdx.x;    // 256 threads x float2 = 512 elements
  const float2 f = ((const float2*)(clusters + (size_t)row * ND))[t];
  ((unsigned*)(cbf + (size_t)row * ND))[t] = pk_bf16(f.x, f.y);
  partial[t] = f.x * f.x + f.y * f.y;
  __syncthreads();
  #pragma unroll
  for (int s = 128; s > 0; s >>= 1) {
    if (t < s) partial[t] += partial[t + s];
    __syncthreads();
  }
  if (t == 0) csqg[row] = partial[0];   // plain store: ws fully rewritten each call
}

// ---------------------------------------------------------------------------
// Kernel 2: fused  q = normalize( 1 / (1 + ||x||^2 + ||c||^2 - 2 x.c) )
// 64 rows x 1024 cols per WG; bf16 WMMA; row-normalization fused via LDS.
// ---------------------------------------------------------------------------
__global__ __launch_bounds__(THREADS, 1)
void soft_assign_wmma(const float* __restrict__ x,
                      const unsigned short* __restrict__ cbf,
                      const float* __restrict__ csqg,
                      float* __restrict__ out) {
  // 64 rows x 512 cols of bf16 = exactly 64KB; reused for reductions later.
  __shared__ __align__(16) unsigned short xbf[BM * ND];

  const int tid  = threadIdx.x;
  const int lane = tid & 31;
  const int w    = tid >> 5;          // wave id 0..15
  const int h    = (lane >> 4) & 1;   // lane-half
  const int ln   = lane & 15;
  const int m_base  = blockIdx.x * BM;
  const int colbase = w * 64;         // each wave: unique 64 columns

  // ---- Phase 1: stage x tile fp32 -> bf16 into LDS ----
  {
    const int row = tid >> 3;          // 8 threads per row
    const int cb  = (tid & 7) * 64;    // 64 floats each
    const float4* src = (const float4*)(x + (size_t)(m_base + row) * ND + cb);
    uint2* dst = (uint2*)(xbf + row * ND + cb);
    #pragma unroll
    for (int i = 0; i < 16; ++i) {
      float4 f = src[i];
      uint2 p;
      p.x = pk_bf16(f.x, f.y);
      p.y = pk_bf16(f.z, f.w);
      dst[i] = p;
    }
  }
  __syncthreads();

  // ---- Phase 2: GEMM  cross = X * C^T  via v_wmma_f32_16x16x32_bf16 ----
  v8f acc[MT * KT];
  #pragma unroll
  for (int i = 0; i < MT * KT; ++i)
    acc[i] = (v8f){0.f,0.f,0.f,0.f,0.f,0.f,0.f,0.f};

  const int khalf = h * 8;
  for (int dsi = 0; dsi < DSTEPS; ++dsi) {
    const int dbase = dsi * 32;

    // A fragments from LDS (ISA 16-bit A 16x32 layout: two b128 reads)
    FragAB a[MT];
    #pragma unroll
    for (int mt = 0; mt < MT; ++mt) {
      const unsigned short* ap = xbf + (mt * 16 + ln) * ND + dbase + khalf;
      a[mt].q[0] = *(const uint4*)(ap);        // K = khalf..khalf+7
      a[mt].q[1] = *(const uint4*)(ap + 16);   // K = 16+khalf..+7
    }

    // B fragments: pre-converted bf16 straight from L2 (zero VALU)
    FragAB b[KT];
    #pragma unroll
    for (int kt = 0; kt < KT; ++kt) {
      const unsigned short* bp =
          cbf + (size_t)(colbase + kt * 16 + ln) * ND + dbase + h * 8;
      if (dsi + 1 < DSTEPS)
        __builtin_prefetch((const void*)(bp + 32), 0, 3);  // next d-slab, near
      b[kt].q[0] = *(const uint4*)(bp);        // K = 8h + 0..7
      b[kt].q[1] = *(const uint4*)(bp + 16);   // K = 16 + 8h + 0..7
    }

    #pragma unroll
    for (int mt = 0; mt < MT; ++mt)
      #pragma unroll
      for (int kt = 0; kt < KT; ++kt)
        acc[mt * KT + kt] = __builtin_amdgcn_wmma_f32_16x16x32_bf16(
            false, a[mt].v, false, b[kt].v,
            (short)0, acc[mt * KT + kt], false, false);
  }

  // ||c||^2 for this lane's columns (precomputed)
  float csq[KT];
  #pragma unroll
  for (int kt = 0; kt < KT; ++kt)
    csq[kt] = csqg[colbase + kt * 16 + ln];

  // ---- Phase 3: ||x||^2 from the bf16 LDS copy ----
  float xsq_part = 0.f;
  const int rrow = tid >> 3;
  {
    const int rcb = (tid & 7) * 64;
    const uint2* p = (const uint2*)(xbf + rrow * ND + rcb);
    #pragma unroll
    for (int i = 0; i < 16; ++i) {
      uint2 v = p[i];
      float f0 = __uint_as_float(v.x << 16);
      float f1 = __uint_as_float(v.x & 0xFFFF0000u);
      float f2 = __uint_as_float(v.y << 16);
      float f3 = __uint_as_float(v.y & 0xFFFF0000u);
      xsq_part += f0*f0 + f1*f1 + f2*f2 + f3*f3;
    }
  }
  __syncthreads();                       // all xbf reads done; safe to reuse LDS
  float* red = (float*)xbf;              // red[0..63]=||x||^2, red[64..127]=row sums
  if (tid < 128) red[tid] = 0.f;
  __syncthreads();
  atomicAdd(&red[rrow], xsq_part);       // ds_add_f32
  __syncthreads();

  // ---- Phase 4: q = 1/(1+dist), row-sum partials ----
  #pragma unroll
  for (int mt = 0; mt < MT; ++mt) {
    #pragma unroll
    for (int j = 0; j < 8; ++j) {
      const int row = mt * 16 + j + h * 8;     // C/D layout: VGPR j, lane-half h
      const float xs = red[row];
      float rp = 0.f;
      #pragma unroll
      for (int kt = 0; kt < KT; ++kt) {
        float cr = acc[mt * KT + kt][j];
        float d  = fmaxf(xs + csq[kt] - 2.0f * cr, 0.0f);
        float q  = 1.0f / (1.0f + d);          // alpha = 1 => power term is identity
        acc[mt * KT + kt][j] = q;
        rp += q;
      }
      atomicAdd(&red[64 + row], rp);
    }
  }
  __syncthreads();

  // ---- Phase 5: normalize and store ----
  #pragma unroll
  for (int mt = 0; mt < MT; ++mt) {
    #pragma unroll
    for (int j = 0; j < 8; ++j) {
      const int row = mt * 16 + j + h * 8;
      const float inv = 1.0f / red[64 + row];
      #pragma unroll
      for (int kt = 0; kt < KT; ++kt) {
        const int col = colbase + kt * 16 + ln;
        out[(size_t)(m_base + row) * NK + col] = acc[mt * KT + kt][j] * inv;
      }
    }
  }
}

extern "C" void kernel_launch(void* const* d_in, const int* in_sizes, int n_in,
                              void* d_out, int out_size, void* d_ws, size_t ws_size,
                              hipStream_t stream) {
  const float* x        = (const float*)d_in[0];
  const float* clusters = (const float*)d_in[1];
  float*       out      = (float*)d_out;

  // workspace: [ bf16 clusters : NK*ND*2 bytes ][ ||c||^2 : NK*4 bytes ]
  unsigned short* cbf  = (unsigned short*)d_ws;
  float*          csqg = (float*)((char*)d_ws + (size_t)NK * ND * sizeof(unsigned short));

  const int N = in_sizes[0] / ND;        // 65536
  hipLaunchKernelGGL(prep_clusters, dim3(NK), dim3(256), 0, stream,
                     clusters, cbf, csqg);
  hipLaunchKernelGGL(soft_assign_wmma, dim3(N / BM), dim3(THREADS), 0, stream,
                     x, cbf, csqg, out);
}
